// SparseGATv2Layer_82712480186382
// MI455X (gfx1250) — matrix operand order repaired and checked by reading, not compiled, not run on hardware
//
#include <hip/hip_runtime.h>
#include <hip/hip_bf16.h>

typedef __attribute__((ext_vector_type(16))) _Float16 v16h;
typedef __attribute__((ext_vector_type(8)))  float    v8f;

#define NEG_SLOPE 0.2f

// ---- order-preserving float<->uint encoding for atomic max ------------------
__device__ __forceinline__ unsigned enc_max(float v) {
    unsigned u = __float_as_uint(v);
    return (u & 0x80000000u) ? ~u : (u | 0x80000000u);
}
__device__ __forceinline__ float dec_max(unsigned k) {
    return (k & 0x80000000u) ? __uint_as_float(k ^ 0x80000000u)
                             : __uint_as_float(~k);
}

// ---- kernel 0: init per-(node,head) max/denominator -------------------------
__global__ void gat_init(unsigned* __restrict__ mx, float* __restrict__ denom, int n) {
    int i = blockIdx.x * blockDim.x + threadIdx.x;
    if (i < n) { mx[i] = 0u; denom[i] = 0.0f; }
}

// ---- kernel 1: fused dual GEMM via WMMA ------------------------------------
// proj = x @ W_lin^T  (to workspace), outr = x @ W_res^T + bias (to d_out)
// fp32 emulated by f16 hi/lo split: A*B ~= Ah*Bh + Ah*Bl + Al*Bh
// Block = 256 threads = 8 waves; wave w computes the 16x16 output tile at
// columns [16w, 16w+16) of output row tile blockIdx.x*16.
//
// Fragment layouts used by this code (A/B: contiguous k order, not the ISA table):
//   A (16x32 f16): lane M = lane&15; elem e    -> K = 16*(lane>>4) + e
//   B (32x16 f16): lane N = lane&15; elem e    -> K = 16*(lane>>4) + e
//   D (16x16 f32): VGPR r -> M = r + 8*(lane>>4), N = lane&15
__global__ __launch_bounds__(256) void gat_gemm(
    const float* __restrict__ x,  const float* __restrict__ Wl,
    const float* __restrict__ Wr, const float* __restrict__ bias,
    float* __restrict__ proj, float* __restrict__ outr, int N)
{
    const int lane = threadIdx.x & 31;
    const int wave = threadIdx.x >> 5;       // 0..7 : column tile
    const int half = lane >> 4;              // 0/1
    const int mn   = lane & 15;
    const int rowbase = blockIdx.x * 16;
    const int colbase = wave * 16;

    int arow = rowbase + mn; if (arow >= N) arow = N - 1;   // clamp (N%16==0 anyway)
    const int wrow = colbase + mn;                          // 0..127, always valid

    v8f accP = {0.f,0.f,0.f,0.f,0.f,0.f,0.f,0.f};
    v8f accR = {0.f,0.f,0.f,0.f,0.f,0.f,0.f,0.f};

    const float* __restrict__ xr  = x  + (size_t)arow * 128;
    const float* __restrict__ wlr = Wl + (size_t)wrow * 128;
    const float* __restrict__ wrr = Wr + (size_t)wrow * 128;

#pragma unroll
    for (int kk = 0; kk < 4; ++kk) {
        const int ka = kk * 32 + 16 * half;   // A: one contiguous run of 16
        const int kb = kk * 32 + 16 * half;   // B: one contiguous run of 16

        float4 a0 = *(const float4*)(xr  + ka);
        float4 a1 = *(const float4*)(xr  + ka + 4);
        float4 a2 = *(const float4*)(xr  + ka + 8);
        float4 a3 = *(const float4*)(xr  + ka + 12);
        float4 l0 = *(const float4*)(wlr + kb);
        float4 l1 = *(const float4*)(wlr + kb + 4);
        float4 l2 = *(const float4*)(wlr + kb + 8);
        float4 l3 = *(const float4*)(wlr + kb + 12);
        float4 r0 = *(const float4*)(wrr + kb);
        float4 r1 = *(const float4*)(wrr + kb + 4);
        float4 r2 = *(const float4*)(wrr + kb + 8);
        float4 r3 = *(const float4*)(wrr + kb + 12);

        float af[16]  = {a0.x,a0.y,a0.z,a0.w, a1.x,a1.y,a1.z,a1.w,
                         a2.x,a2.y,a2.z,a2.w, a3.x,a3.y,a3.z,a3.w};
        float blf[16] = {l0.x,l0.y,l0.z,l0.w, l1.x,l1.y,l1.z,l1.w,
                         l2.x,l2.y,l2.z,l2.w, l3.x,l3.y,l3.z,l3.w};
        float brf[16] = {r0.x,r0.y,r0.z,r0.w, r1.x,r1.y,r1.z,r1.w,
                         r2.x,r2.y,r2.z,r2.w, r3.x,r3.y,r3.z,r3.w};

        v16h a_hi, a_lo, bl_hi, bl_lo, br_hi, br_lo;
#pragma unroll
        for (int e = 0; e < 16; ++e) {
            _Float16 h;
            h = (_Float16)af[e];  a_hi[e]  = h; a_lo[e]  = (_Float16)(af[e]  - (float)h);
            h = (_Float16)blf[e]; bl_hi[e] = h; bl_lo[e] = (_Float16)(blf[e] - (float)h);
            h = (_Float16)brf[e]; br_hi[e] = h; br_lo[e] = (_Float16)(brf[e] - (float)h);
        }

        accP = __builtin_amdgcn_wmma_f32_16x16x32_f16(false, a_hi, false, bl_hi, (short)0, accP, false, false);
        accP = __builtin_amdgcn_wmma_f32_16x16x32_f16(false, a_hi, false, bl_lo, (short)0, accP, false, false);
        accP = __builtin_amdgcn_wmma_f32_16x16x32_f16(false, a_lo, false, bl_hi, (short)0, accP, false, false);

        accR = __builtin_amdgcn_wmma_f32_16x16x32_f16(false, a_hi, false, br_hi, (short)0, accR, false, false);
        accR = __builtin_amdgcn_wmma_f32_16x16x32_f16(false, a_hi, false, br_lo, (short)0, accR, false, false);
        accR = __builtin_amdgcn_wmma_f32_16x16x32_f16(false, a_lo, false, br_hi, (short)0, accR, false, false);
    }

    const float b = bias[colbase + mn];
#pragma unroll
    for (int r = 0; r < 8; ++r) {
        int row = rowbase + 8 * half + r;           // D layout: VGPR r -> M = r + 8*half
        if (row < N) {
            size_t idx = (size_t)row * 128 + colbase + mn;
            proj[idx] = accP[r];
            outr[idx] = accR[r] + b;
        }
    }
}

// ---- kernel 2: per-edge logits + segment max (wave per edge) ----------------
__global__ __launch_bounds__(256) void gat_edge_logits(
    const int* __restrict__ ei, const float* __restrict__ ea,
    const float* __restrict__ proj, const float* __restrict__ att,
    const float* __restrict__ web, float* __restrict__ logits,
    unsigned* __restrict__ mx, int E, int Et)
{
    int gid  = blockIdx.x * blockDim.x + threadIdx.x;
    int lane = gid & 31;
    int e    = gid >> 5;
    if (e >= Et) return;

    int src, dst; float red;
    if (e < E) { src = ei[e]; dst = ei[E + e]; red = ea[2*e] + 0.35f * ea[2*e + 1]; }
    else       { src = e - E; dst = src;       red = 1.0f; }

    const float* ps = proj + (size_t)src * 128;
    const float* pd = proj + (size_t)dst * 128;

    float lv[4];
#pragma unroll
    for (int h = 0; h < 4; ++h) {
        float z = ps[h*32 + lane] + pd[h*32 + lane];
        z = (z > 0.0f) ? z : NEG_SLOPE * z;
        lv[h] = z * att[h*32 + lane];
    }
#pragma unroll
    for (int off = 16; off > 0; off >>= 1) {
#pragma unroll
        for (int h = 0; h < 4; ++h) lv[h] += __shfl_xor(lv[h], off, 32);
    }
    if (lane == 0) {
#pragma unroll
        for (int h = 0; h < 4; ++h) {
            float val = lv[h] + red * web[h];
            logits[(size_t)e * 4 + h] = val;
            atomicMax(mx + (size_t)dst * 4 + h, enc_max(val));
        }
    }
}

// ---- kernel 3: exp(logit - max) + segment sum (thread per edge-head) --------
__global__ void gat_exp(const int* __restrict__ ei, float* __restrict__ logits,
                        const unsigned* __restrict__ mx, float* __restrict__ denom,
                        int E, int Et)
{
    int tid = blockIdx.x * blockDim.x + threadIdx.x;
    if (tid >= Et * 4) return;
    int e = tid >> 2, h = tid & 3;
    int dst = (e < E) ? ei[E + e] : (e - E);
    float m = dec_max(mx[(size_t)dst * 4 + h]);
    float v = expf(logits[tid] - m);
    logits[tid] = v;
    atomicAdd(denom + (size_t)dst * 4 + h, v);
}

// ---- kernel 4: normalize + scatter-add messages (wave per edge) -------------
__global__ __launch_bounds__(256) void gat_scatter(
    const int* __restrict__ ei, const float* __restrict__ logits,
    const float* __restrict__ denom, const float* __restrict__ proj,
    float* __restrict__ out, int E, int Et)
{
    int gid  = blockIdx.x * blockDim.x + threadIdx.x;
    int lane = gid & 31;
    int e    = gid >> 5;
    if (e >= Et) return;

    int src, dst;
    if (e < E) { src = ei[e]; dst = ei[E + e]; }
    else       { src = e - E; dst = src; }

    const float* ps = proj + (size_t)src * 128;
    float*       od = out  + (size_t)dst * 128;
#pragma unroll
    for (int h = 0; h < 4; ++h) {
        float a = logits[(size_t)e * 4 + h] / fmaxf(denom[(size_t)dst * 4 + h], 1e-12f);
        atomicAdd(od + h*32 + lane, a * ps[h*32 + lane]);
    }
}

extern "C" void kernel_launch(void* const* d_in, const int* in_sizes, int n_in,
                              void* d_out, int out_size, void* d_ws, size_t ws_size,
                              hipStream_t stream) {
    const float* x    = (const float*)d_in[0];
    const int*   ei   = (const int*)  d_in[1];
    const float* ea   = (const float*)d_in[2];
    const float* Wl   = (const float*)d_in[3];
    const float* att  = (const float*)d_in[4];
    const float* web  = (const float*)d_in[5];
    const float* bias = (const float*)d_in[6];
    const float* Wr   = (const float*)d_in[7];
    float* out = (float*)d_out;

    const int N  = in_sizes[0] / 128;
    const int E  = in_sizes[1] / 2;
    const int Et = E + N;

    char* ws = (char*)d_ws;
    float*    proj   = (float*)ws;    ws += (size_t)N  * 128 * sizeof(float);
    float*    logits = (float*)ws;    ws += (size_t)Et * 4   * sizeof(float);
    unsigned* mx     = (unsigned*)ws; ws += (size_t)N  * 4   * sizeof(unsigned);
    float*    denom  = (float*)ws;

    gat_init<<<(N*4 + 255)/256, 256, 0, stream>>>(mx, denom, N*4);
    gat_gemm<<<(N + 15)/16, 256, 0, stream>>>(x, Wl, Wr, bias, proj, out, N);

    long long ethreads = (long long)Et * 32;
    gat_edge_logits<<<(unsigned)((ethreads + 255)/256), 256, 0, stream>>>(
        ei, ea, proj, att, web, logits, mx, E, Et);
    gat_exp<<<(Et*4 + 255)/256, 256, 0, stream>>>(ei, logits, mx, denom, E, Et);
    gat_scatter<<<(unsigned)((ethreads + 255)/256), 256, 0, stream>>>(
        ei, logits, denom, proj, out, E, Et);
}
